// GatedDeltaNet_34832184770896
// MI455X (gfx1250) — compile-verified, matrix operand order checked by reading, not run on hardware
//
#include <hip/hip_runtime.h>
#include <hip/hip_bf16.h>
#include <cstdint>
#include <cstddef>

// ---------------------------------------------------------------------------
// GatedDeltaNet forward for MI455X (gfx1250), bf16 WMMA everywhere,
// async global->LDS staging (ASYNCcnt) where the toolchain supports it.
// ---------------------------------------------------------------------------

typedef __attribute__((ext_vector_type(16))) __bf16 v16bf;
typedef __attribute__((ext_vector_type(8)))  float  v8f;
typedef __attribute__((ext_vector_type(4)))  int    v4i;

#define DEV __device__ __forceinline__

constexpr int BATCH   = 2;
constexpr int SEQ     = 4096;
constexpr int HID     = 2048;
constexpr int NKH     = 16;
constexpr int NVH     = 32;
constexpr int HK      = 128;
constexpr int HV      = 128;
constexpr int KEY_DIM = NKH * HK;          // 2048
constexpr int VAL_DIM = NVH * HV;          // 4096
constexpr int CONVD   = 2 * KEY_DIM + VAL_DIM; // 8192
constexpr int ROWS    = BATCH * SEQ;       // 8192
constexpr int CHUNK   = 64;
constexpr int NCHUNK  = SEQ / CHUNK;       // 64
constexpr int VSPLIT  = 2;                 // split V=128 into 2 halves of 64
constexpr int HVS     = HV / VSPLIT;       // 64

// ---------------------- async global->LDS support --------------------------

#if defined(__has_builtin)
#  if __has_builtin(__builtin_amdgcn_global_load_async_to_lds_b128)
#    define HAVE_ASYNC_LDS 1
#  endif
#endif
#ifndef HAVE_ASYNC_LDS
#  define HAVE_ASYNC_LDS 0
#endif

#if HAVE_ASYNC_LDS
// Toolchain declares the pointer params as `int __vector(4) *` (see round-2
// diagnostic), so cast to plain v4i* and let the backend resolve addrspaces.
DEV void async_copy_b128(const void* gsrc, void* ldst) {
  __builtin_amdgcn_global_load_async_to_lds_b128(
      (v4i*)gsrc, (v4i*)ldst, 0, 0);
}
DEV void wait_async_all() {
#  if __has_builtin(__builtin_amdgcn_s_wait_asynccnt)
  __builtin_amdgcn_s_wait_asynccnt(0);
#  else
  asm volatile("s_wait_asynccnt 0" ::: "memory");
#  endif
}
#endif

// -------------------------- WMMA helpers -----------------------------------

DEV v8f wmma_bf16(v16bf a, v16bf b, v8f c) {
  return __builtin_amdgcn_wmma_f32_16x16x32_bf16(false, a, false, b,
                                                 (short)0, c, false, false);
}

// A fragment: 16x32 tile at (tm, kk). atr => logical A(m,k) = A[k*lda + m].
// ISA layout: row = lane%16; lane<16 holds K {0..7,16..23}, lane>=16 {8..15,24..31}.
DEV v16bf frag_A(const __bf16* A, int lda, int tm, int kk, int lane, bool atr) {
  v16bf r;
  int row  = lane & 15;
  int half = (lane >> 4) * 8;
#pragma unroll
  for (int e = 0; e < 16; ++e) {
    int k  = (e < 8) ? (half + e) : (16 + half + (e - 8));
    int m  = tm + row;
    int kq = kk + k;
    r[e] = atr ? A[(size_t)kq * lda + m] : A[(size_t)m * lda + kq];
  }
  return r;
}

// B fragment: 32x16 tile at (kk, tn). btr => logical B(k,n) = B[n*ldb + k].
DEV v16bf frag_B(const __bf16* B, int ldb, int kk, int tn, int lane, bool btr) {
  v16bf r;
  int col  = lane & 15;
  int half = (lane >> 4) * 8;
#pragma unroll
  for (int e = 0; e < 16; ++e) {
    int k  = (e < 8) ? (half + e) : (16 + half + (e - 8));
    int kq = kk + k;
    r[e] = btr ? B[(size_t)(tn + col) * ldb + kq]
               : B[(size_t)kq * ldb + tn + col];
  }
  return r;
}

// Block-cooperative GEMM on LDS/global operands (256 threads = 8 waves).
// D = scale * op(A)*op(B) + Cin ; write fp32 to Fout and/or bf16 to Bout.
DEV void bgemm(const __bf16* A, int lda, bool atr,
               const __bf16* B, int ldb, bool btr,
               const float* Cin, int ldci, float scale,
               float* Fout, __bf16* Bout, int ldo,
               int M, int N, int K) {
  int tid  = threadIdx.x;
  int lane = tid & 31;
  int wave = tid >> 5;
  int ntn  = N >> 4;
  int tiles = (M >> 4) * ntn;
  for (int t = wave; t < tiles; t += 8) {
    int tm = (t / ntn) << 4;
    int tn = (t % ntn) << 4;
    v8f acc;
#pragma unroll
    for (int i = 0; i < 8; ++i) acc[i] = 0.f;
    for (int kk = 0; kk < K; kk += 32) {
      v16bf a = frag_A(A, lda, tm, kk, lane, atr);
      v16bf b = frag_B(B, ldb, kk, tn, lane, btr);
      acc = wmma_bf16(a, b, acc);
    }
    int half = lane >> 4, col = lane & 15;
#pragma unroll
    for (int i = 0; i < 8; ++i) {
      int m = tm + i + half * 8;
      int n = tn + col;
      float vv = scale * acc[i];
      if (Cin)  vv += Cin[(size_t)m * ldci + n];
      if (Fout) Fout[(size_t)m * ldo + n] = vv;
      if (Bout) Bout[(size_t)m * ldo + n] = (__bf16)vv;
    }
  }
}

// -------------------------- cast kernel ------------------------------------

__global__ void cast_f32_bf16_kernel(const float* __restrict__ src,
                                     __bf16* __restrict__ dst, int n) {
  int i = blockIdx.x * blockDim.x + threadIdx.x;
  if (i < n) dst[i] = (__bf16)src[i];
}

// -------------------------- big tiled GEMM ---------------------------------
// C[M,N] = A[M,K](bf16) * B[K,N](bf16); 128x128 block, BK=32, 8 waves x 32x64.
// Double-buffered LDS with async global->LDS staging (tile t+1 in flight while
// WMMA consumes tile t).

template <bool OUT_BF>
__global__ __launch_bounds__(256) void gemm_kernel(
    const __bf16* __restrict__ A, const __bf16* __restrict__ B,
    void* __restrict__ Cv, int M, int N, int K) {
  __shared__ __bf16 As[2][128][32];
  __shared__ __bf16 Bs[2][32][128];
  int bm = blockIdx.x * 128, bn = blockIdx.y * 128;
  int tid = threadIdx.x, lane = tid & 31, wave = tid >> 5;
  int wm = (wave & 3) * 32;
  int wn = (wave >> 2) * 64;

  v8f acc[2][4];
#pragma unroll
  for (int i = 0; i < 2; ++i)
#pragma unroll
    for (int j = 0; j < 4; ++j)
#pragma unroll
      for (int e = 0; e < 8; ++e) acc[i][j][e] = 0.f;

  auto stage = [&](int buf, int k0) {
#pragma unroll
    for (int i = 0; i < 2; ++i) {
      int vid = tid + i * 256;                 // 0..511
      int ar = vid >> 2, ac = (vid & 3) * 8;   // A: 128 rows x 4 vec8
      int br = vid >> 4, bc = (vid & 15) * 8;  // B: 32 rows x 16 vec8
      const __bf16* ga = A + (size_t)(bm + ar) * K + k0 + ac;
      const __bf16* gb = B + (size_t)(k0 + br) * N + bn + bc;
#if HAVE_ASYNC_LDS
      async_copy_b128(ga, &As[buf][ar][ac]);
      async_copy_b128(gb, &Bs[buf][br][bc]);
#else
      *reinterpret_cast<uint4*>(&As[buf][ar][ac]) =
          *reinterpret_cast<const uint4*>(ga);
      *reinterpret_cast<uint4*>(&Bs[buf][br][bc]) =
          *reinterpret_cast<const uint4*>(gb);
      if (k0 + 32 < K) {  // gfx1250 global_prefetch_b8
        __builtin_prefetch(ga + 32, 0, 1);
        __builtin_prefetch(gb + (size_t)32 * N, 0, 1);
      }
#endif
    }
  };

  stage(0, 0);
#if HAVE_ASYNC_LDS
  wait_async_all();
#endif
  __syncthreads();

  int cur = 0;
  for (int k0 = 0; k0 < K; k0 += 32) {
    if (k0 + 32 < K) stage(cur ^ 1, k0 + 32);  // overlap DMA with WMMA

    v16bf af[2], bfv[4];
#pragma unroll
    for (int i = 0; i < 2; ++i)
      af[i] = frag_A(&As[cur][0][0], 32, wm + i * 16, 0, lane, false);
#pragma unroll
    for (int j = 0; j < 4; ++j)
      bfv[j] = frag_B(&Bs[cur][0][0], 128, 0, wn + j * 16, lane, false);
#pragma unroll
    for (int i = 0; i < 2; ++i)
#pragma unroll
      for (int j = 0; j < 4; ++j)
        acc[i][j] = wmma_bf16(af[i], bfv[j], acc[i][j]);

#if HAVE_ASYNC_LDS
    wait_async_all();
#endif
    __syncthreads();
    cur ^= 1;
  }

  int half = lane >> 4, col = lane & 15;
#pragma unroll
  for (int i = 0; i < 2; ++i)
#pragma unroll
    for (int j = 0; j < 4; ++j)
#pragma unroll
      for (int e = 0; e < 8; ++e) {
        int m = bm + wm + i * 16 + e + half * 8;
        int n = bn + wn + j * 16 + col;
        if (OUT_BF)
          ((__bf16*)Cv)[(size_t)m * N + n] = (__bf16)acc[i][j][e];
        else
          ((float*)Cv)[(size_t)m * N + n] = acc[i][j][e];
      }
}

// -------------------------- beta / g projection ----------------------------

__global__ __launch_bounds__(256) void beta_g_kernel(
    const float* __restrict__ x, const float* __restrict__ wb,
    const float* __restrict__ wa, const float* __restrict__ dt_bias,
    const float* __restrict__ a_log, float* __restrict__ beta,
    float* __restrict__ g) {
  __shared__ float xs[HID];
  __shared__ float part[256];
  int r = blockIdx.x, tid = threadIdx.x;
  for (int i = tid; i < HID; i += 256) xs[i] = x[(size_t)r * HID + i];
  __syncthreads();
  int pair = tid >> 2, seg = tid & 3;  // 64 dot products, 4 threads each
  int h = pair & 31;
  bool isA = pair >= 32;
  const float* w = isA ? wa : wb;
  float s = 0.f;
  int j0 = seg * (HID / 4);
  for (int j = j0; j < j0 + HID / 4; ++j) s += xs[j] * w[(size_t)j * NVH + h];
  part[tid] = s;
  __syncthreads();
  if (seg == 0) {
    float d = part[tid] + part[tid + 1] + part[tid + 2] + part[tid + 3];
    if (!isA) {
      beta[(size_t)r * NVH + h] = 1.f / (1.f + __expf(-d));
    } else {
      float t = d + dt_bias[h];
      float sp = (t > 20.f) ? t : log1pf(__expf(t));
      g[(size_t)r * NVH + h] = -__expf(a_log[h]) * sp;
    }
  }
}

// --------------- depthwise conv + silu (+ l2norm for q/k) ------------------

__global__ __launch_bounds__(128) void conv_qk_kernel(
    const __bf16* __restrict__ qkv, const float* __restrict__ conv_w,
    __bf16* __restrict__ qo, __bf16* __restrict__ ko) {
  int r = blockIdx.x;        // global row
  int head = blockIdx.y;     // 0..15 q heads, 16..31 k heads
  int t = threadIdx.x;       // channel in head
  int s = r & (SEQ - 1);
  bool isq = head < NKH;
  int hh = isq ? head : head - NKH;
  int d = (isq ? 0 : KEY_DIM) + hh * HK + t;
  float acc = 0.f;
#pragma unroll
  for (int j = 0; j < 4; ++j) {
    int ss = s - 3 + j;
    if (ss >= 0)
      acc += (float)qkv[(size_t)(r - 3 + j) * CONVD + d] * conv_w[d * 4 + j];
  }
  float xv = acc / (1.f + __expf(-acc));  // silu
  __shared__ float red[128];
  red[t] = xv * xv;
  __syncthreads();
  for (int o = 64; o > 0; o >>= 1) {
    if (t < o) red[t] += red[t + o];
    __syncthreads();
  }
  float outv = xv * rsqrtf(red[0] + 1e-6f);
  if (isq) {
    outv *= 0.08838834764831845f;  // K^-0.5, K=128
    qo[(size_t)r * KEY_DIM + hh * HK + t] = (__bf16)outv;
  } else {
    ko[(size_t)r * KEY_DIM + hh * HK + t] = (__bf16)outv;
  }
}

__global__ __launch_bounds__(256) void conv_v_kernel(
    const __bf16* __restrict__ qkv, const float* __restrict__ conv_w,
    __bf16* __restrict__ vo) {
  size_t idx = (size_t)blockIdx.x * 256 + threadIdx.x;  // < ROWS*VAL_DIM
  int r = (int)(idx >> 12);
  int c = (int)(idx & (VAL_DIM - 1));
  int s = r & (SEQ - 1);
  int d = 2 * KEY_DIM + c;
  float acc = 0.f;
#pragma unroll
  for (int j = 0; j < 4; ++j) {
    int ss = s - 3 + j;
    if (ss >= 0)
      acc += (float)qkv[(size_t)(r - 3 + j) * CONVD + d] * conv_w[d * 4 + j];
  }
  vo[idx] = (__bf16)(acc / (1.f + __expf(-acc)));
}

// ------------------------ core: gated delta rule ---------------------------
// One workgroup per (b, value-head, V-half). Sequential scan over 64 chunks,
// state (HKxHVS fp32 + bf16 shadow) resident in LDS; all matmuls via WMMA.

constexpr int SM_STATE  = 0;                          // f32 128*64
constexpr int SM_STATEB = SM_STATE + HK * HVS * 4;    // bf16 128*64
constexpr int SM_QS     = SM_STATEB + HK * HVS * 2;   // bf16 64*128
constexpr int SM_KS     = SM_QS  + CHUNK * HK * 2;
constexpr int SM_KB     = SM_KS  + CHUNK * HK * 2;    // kbeta / later kdecay
constexpr int SM_KE     = SM_KB  + CHUNK * HK * 2;    // kbeta*exp(g)
constexpr int SM_KCD    = SM_KE  + CHUNK * HK * 2;    // attn@(kbeta*exp(g))
constexpr int SM_VB     = SM_KCD + CHUNK * HK * 2;    // bf16 64*64
constexpr int SM_VNEW   = SM_VB    + CHUNK * HVS * 2;
constexpr int SM_ATTNB  = SM_VNEW  + CHUNK * HVS * 2; // bf16 64*64
constexpr int SM_ATTN2  = SM_ATTNB + CHUNK * CHUNK * 2;
constexpr int SM_ATTNF  = SM_ATTN2 + CHUNK * CHUNK * 2; // f32 64*64
constexpr int SM_XSOL   = SM_ATTNF + CHUNK * CHUNK * 4;
constexpr int SM_VC     = SM_XSOL  + CHUNK * CHUNK * 4;
constexpr int SM_TMPF   = SM_VC    + CHUNK * HVS * 4;
constexpr int SM_GC     = SM_TMPF  + CHUNK * HVS * 4;  // f32 64
constexpr int SM_BETA   = SM_GC + CHUNK * 4;
constexpr int CORE_SMEM = SM_BETA + CHUNK * 4;         // ~225 KB

__global__ __launch_bounds__(256) void core_kernel(
    const __bf16* __restrict__ q, const __bf16* __restrict__ k,
    const __bf16* __restrict__ v, const float* __restrict__ beta,
    const float* __restrict__ g, float* __restrict__ out) {
  extern __shared__ char smem[];
  float*  state  = (float*)(smem + SM_STATE);
  __bf16* stateb = (__bf16*)(smem + SM_STATEB);
  __bf16* qs     = (__bf16*)(smem + SM_QS);
  __bf16* ks     = (__bf16*)(smem + SM_KS);
  __bf16* kb     = (__bf16*)(smem + SM_KB);
  __bf16* ke     = (__bf16*)(smem + SM_KE);
  __bf16* kcd    = (__bf16*)(smem + SM_KCD);
  __bf16* vb     = (__bf16*)(smem + SM_VB);
  __bf16* vnew   = (__bf16*)(smem + SM_VNEW);
  __bf16* attnb  = (__bf16*)(smem + SM_ATTNB);
  __bf16* attn2  = (__bf16*)(smem + SM_ATTN2);
  float*  attnf  = (float*)(smem + SM_ATTNF);
  float*  xsol   = (float*)(smem + SM_XSOL);
  float*  vc     = (float*)(smem + SM_VC);
  float*  tmpf   = (float*)(smem + SM_TMPF);
  float*  gc     = (float*)(smem + SM_GC);
  float*  betas  = (float*)(smem + SM_BETA);

  int bh = blockIdx.x >> 1;      // b*NVH + h
  int vh = blockIdx.x & 1;       // V half
  int b = bh >> 5, h = bh & 31;
  int kh = h >> 1;               // GQA repeat: key head = h/2
  int tid = threadIdx.x;

  for (int i = tid; i < HK * HVS; i += 256) {
    state[i] = 0.f;
    stateb[i] = (__bf16)0.f;
  }
  __syncthreads();

  int row00 = b * SEQ;
  for (int ch = 0; ch < NCHUNK; ++ch) {
    int r0 = row00 + ch * CHUNK;

    // ---- stage raw q,k,v for this chunk (async DMA overlaps cumsum) ----
#if HAVE_ASYNC_LDS
    for (int vid = tid; vid < CHUNK * (HK / 8); vid += 256) {   // 1024 vec8
      int r = vid >> 4, c = (vid & 15) * 8;
      async_copy_b128(q + (size_t)(r0 + r) * KEY_DIM + kh * HK + c,
                      &qs[r * HK + c]);
      async_copy_b128(k + (size_t)(r0 + r) * KEY_DIM + kh * HK + c,
                      &ks[r * HK + c]);
    }
    for (int vid = tid; vid < CHUNK * (HVS / 8); vid += 256) {  // 512 vec8
      int r = vid >> 3, c = (vid & 7) * 8;
      async_copy_b128(v + (size_t)(r0 + r) * VAL_DIM + h * HV + vh * HVS + c,
                      &vb[r * HVS + c]);
    }
#endif
    if (tid < CHUNK) {
      gc[tid]    = g[(size_t)(r0 + tid) * NVH + h];
      betas[tid] = beta[(size_t)(r0 + tid) * NVH + h];
    }
    __syncthreads();
    if (tid == 0) {  // inclusive cumsum of g over the chunk
      float s = 0.f;
      for (int i = 0; i < CHUNK; ++i) { s += gc[i]; gc[i] = s; }
    }
#if HAVE_ASYNC_LDS
    wait_async_all();
#endif
    __syncthreads();

    // derive k*beta, k*beta*exp(g), v*beta (in LDS)
#if HAVE_ASYNC_LDS
    for (int idx = tid; idx < CHUNK * HK; idx += 256) {
      int r = idx >> 7;
      float kf = (float)ks[idx], bt = betas[r];
      kb[idx] = (__bf16)(kf * bt);
      ke[idx] = (__bf16)(kf * bt * __expf(gc[r]));
    }
    for (int idx = tid; idx < CHUNK * HVS; idx += 256) {
      int r = idx >> 6;
      vb[idx] = (__bf16)((float)vb[idx] * betas[r]);
    }
#else
    for (int idx = tid; idx < CHUNK * HK; idx += 256) {
      int r = idx >> 7, c = idx & 127;
      __bf16 qv = q[(size_t)(r0 + r) * KEY_DIM + kh * HK + c];
      __bf16 kv = k[(size_t)(r0 + r) * KEY_DIM + kh * HK + c];
      qs[idx] = qv;
      ks[idx] = kv;
      float kf = (float)kv, bt = betas[r];
      kb[idx] = (__bf16)(kf * bt);
      ke[idx] = (__bf16)(kf * bt * __expf(gc[r]));
    }
    for (int idx = tid; idx < CHUNK * HVS; idx += 256) {
      int r = idx >> 6, c = idx & 63;
      float vv = (float)v[(size_t)(r0 + r) * VAL_DIM + h * HV + vh * HVS + c];
      vb[idx] = (__bf16)(vv * betas[r]);
    }
#endif
    __syncthreads();

    // attnf = -(k_beta @ k^T)
    bgemm(kb, HK, false, ks, HK, true, nullptr, 0, -1.f,
          attnf, nullptr, CHUNK, CHUNK, CHUNK, HK);
    __syncthreads();
    // A = strict_lower ? attnf * exp(gc_i - gc_j) : 0
    for (int idx = tid; idx < CHUNK * CHUNK; idx += 256) {
      int i = idx >> 6, j = idx & 63;
      attnf[idx] = (i > j) ? attnf[idx] * __expf(gc[i] - gc[j]) : 0.f;
    }
    __syncthreads();
    // X = (I - A)^-1, unit lower; each thread owns one column
    if (tid < CHUNK) {
      int c = tid;
      for (int i = 0; i < CHUNK; ++i) {
        float s = (i == c) ? 1.f : 0.f;
        for (int j = 0; j < i; ++j) s += attnf[i * CHUNK + j] * xsol[j * CHUNK + c];
        xsol[i * CHUNK + c] = s;
      }
    }
    __syncthreads();
    for (int idx = tid; idx < CHUNK * CHUNK; idx += 256)
      attnb[idx] = (__bf16)xsol[idx];
    __syncthreads();

    // vc = attn @ v_beta ; kcd = attn @ (k_beta*exp(g))
    bgemm(attnb, CHUNK, false, vb, HVS, false, nullptr, 0, 1.f,
          vc, nullptr, HVS, CHUNK, HVS, CHUNK);
    bgemm(attnb, CHUNK, false, ke, HK, false, nullptr, 0, 1.f,
          nullptr, kcd, HK, CHUNK, HK, CHUNK);
    __syncthreads();

    // attn2 = lower ? (q @ k^T) * decay : 0   (reuse attnf)
    bgemm(qs, HK, false, ks, HK, true, nullptr, 0, 1.f,
          attnf, nullptr, CHUNK, CHUNK, CHUNK, HK);
    __syncthreads();
    for (int idx = tid; idx < CHUNK * CHUNK; idx += 256) {
      int i = idx >> 6, j = idx & 63;
      attn2[idx] = (__bf16)((i >= j) ? attnf[idx] * __expf(gc[i] - gc[j]) : 0.f);
    }
    __syncthreads();

    // v_new = vc - kcd @ state
    bgemm(kcd, HK, false, stateb, HVS, false, vc, HVS, -1.f,
          nullptr, vnew, HVS, CHUNK, HVS, HK);
    __syncthreads();

    // q <- q * exp(gc) (in place; raw q no longer needed)
    for (int idx = tid; idx < CHUNK * HK; idx += 256) {
      int r = idx >> 7;
      qs[idx] = (__bf16)((float)qs[idx] * __expf(gc[r]));
    }
    __syncthreads();

    // tmpf = (q*exp(g)) @ state
    bgemm(qs, HK, false, stateb, HVS, false, nullptr, 0, 1.f,
          tmpf, nullptr, HVS, CHUNK, HVS, HK);
    __syncthreads();

    // o = attn2 @ v_new + tmpf  -> global fp32 core output
    float* optr = out + (size_t)r0 * VAL_DIM + h * HV + vh * HVS;
    bgemm(attn2, CHUNK, false, vnew, HVS, false, tmpf, HVS, 1.f,
          optr, nullptr, VAL_DIM, CHUNK, HVS, CHUNK);
    __syncthreads();

    // state = state*exp(g_last) + (k * exp(g_last - g))^T @ v_new
    float gl = gc[CHUNK - 1];
    for (int idx = tid; idx < CHUNK * HK; idx += 256) {
      int r = idx >> 7;
      kb[idx] = (__bf16)((float)ks[idx] * __expf(gl - gc[r]));  // kdecay
    }
    float sc = __expf(gl);
    for (int idx = tid; idx < HK * HVS; idx += 256) state[idx] *= sc;
    __syncthreads();
    bgemm(kb, HK, true, vnew, HVS, false, state, HVS, 1.f,
          state, nullptr, HVS, HK, HVS, CHUNK);
    __syncthreads();
    for (int idx = tid; idx < HK * HVS; idx += 256)
      stateb[idx] = (__bf16)state[idx];
    __syncthreads();
  }
}

// ------------------------ gated RMSNorm ------------------------------------

__global__ __launch_bounds__(128) void norm_gate_kernel(
    const float* __restrict__ core, const __bf16* __restrict__ z,
    const float* __restrict__ norm_w, __bf16* __restrict__ y) {
  int t = threadIdx.x;
  size_t base = (size_t)blockIdx.x * HV;  // (r*NVH + h) * 128 == r*VAL_DIM + h*128
  float c = core[base + t];
  __shared__ float red[128];
  red[t] = c * c;
  __syncthreads();
  for (int o = 64; o > 0; o >>= 1) {
    if (t < o) red[t] += red[t + o];
    __syncthreads();
  }
  float inv = rsqrtf(red[0] * (1.f / HV) + 1e-6f);
  float zf = (float)z[base + t];
  float silu = zf / (1.f + __expf(-zf));
  y[base + t] = (__bf16)(c * inv * norm_w[t] * silu);
}

// ------------------------------ host ---------------------------------------

static inline size_t align256(size_t x) { return (x + 255) & ~(size_t)255; }

extern "C" void kernel_launch(void* const* d_in, const int* in_sizes, int n_in,
                              void* d_out, int out_size, void* d_ws, size_t ws_size,
                              hipStream_t stream) {
  (void)in_sizes; (void)n_in; (void)out_size; (void)ws_size;
  const float* x       = (const float*)d_in[0];
  const float* w_qkv   = (const float*)d_in[1];
  const float* w_z     = (const float*)d_in[2];
  const float* w_b     = (const float*)d_in[3];
  const float* w_a     = (const float*)d_in[4];
  const float* conv_w  = (const float*)d_in[5];
  const float* dt_bias = (const float*)d_in[6];
  const float* a_log   = (const float*)d_in[7];
  const float* norm_w  = (const float*)d_in[8];
  const float* w_out   = (const float*)d_in[9];
  float* out = (float*)d_out;

  char* ws = (char*)d_ws;
  size_t off = 0;
  auto carve = [&](size_t bytes) { size_t r = off; off += align256(bytes); return r; };

  __bf16* xbf    = (__bf16*)(ws + carve((size_t)ROWS * HID * 2));
  __bf16* wqkvb  = (__bf16*)(ws + carve((size_t)HID * CONVD * 2));
  __bf16* wzb    = (__bf16*)(ws + carve((size_t)HID * VAL_DIM * 2));
  __bf16* woutb  = (__bf16*)(ws + carve((size_t)VAL_DIM * HID * 2));
  __bf16* qkvb   = (__bf16*)(ws + carve((size_t)ROWS * CONVD * 2));
  __bf16* zb     = (__bf16*)(ws + carve((size_t)ROWS * VAL_DIM * 2));
  float*  betaf  = (float*)(ws + carve((size_t)ROWS * NVH * 4));
  float*  gf     = (float*)(ws + carve((size_t)ROWS * NVH * 4));
  __bf16* qb     = (__bf16*)(ws + carve((size_t)ROWS * KEY_DIM * 2));
  __bf16* kb     = (__bf16*)(ws + carve((size_t)ROWS * KEY_DIM * 2));
  __bf16* vb     = (__bf16*)(ws + carve((size_t)ROWS * VAL_DIM * 2));
  // Aliases (safe: stream-ordered kernel sequence):
  float*  coref  = (float*)qkvb;   // core fp32 output reuses qkv buffer (128 MiB each)
  __bf16* yb     = xbf;            // gated output reuses xbf+wqkvb region (64 MiB)

  // 1) fp32 -> bf16 casts
  auto cast = [&](const float* s, __bf16* d, int n) {
    cast_f32_bf16_kernel<<<(n + 255) / 256, 256, 0, stream>>>(s, d, n);
  };
  cast(x,     xbf,   ROWS * HID);
  cast(w_qkv, wqkvb, HID * CONVD);
  cast(w_z,   wzb,   HID * VAL_DIM);
  cast(w_out, woutb, VAL_DIM * HID);

  // 2) qkv = x @ w_qkv   (8192 x 8192 x 2048)
  gemm_kernel<true><<<dim3(ROWS / 128, CONVD / 128), 256, 0, stream>>>(
      xbf, wqkvb, qkvb, ROWS, CONVD, HID);
  // 3) z = x @ w_z       (8192 x 4096 x 2048)
  gemm_kernel<true><<<dim3(ROWS / 128, VAL_DIM / 128), 256, 0, stream>>>(
      xbf, wzb, zb, ROWS, VAL_DIM, HID);

  // 4) beta / g projections
  beta_g_kernel<<<ROWS, 256, 0, stream>>>(x, w_b, w_a, dt_bias, a_log, betaf, gf);

  // 5) depthwise conv + silu (+ l2norm & scale for q/k)
  conv_qk_kernel<<<dim3(ROWS, 2 * NKH), 128, 0, stream>>>(qkvb, conv_w, qb, kb);
  conv_v_kernel<<<(ROWS * VAL_DIM) / 256, 256, 0, stream>>>(qkvb, conv_w, vb);

  // 6) chunked gated delta rule scan (writes coref, aliasing qkvb)
  (void)hipFuncSetAttribute((const void*)core_kernel,
                            hipFuncAttributeMaxDynamicSharedMemorySize,
                            (int)CORE_SMEM);
  core_kernel<<<BATCH * NVH * VSPLIT, 256, CORE_SMEM, stream>>>(
      qb, kb, vb, betaf, gf, coref);

  // 7) gated RMSNorm -> yb (aliasing xbf region)
  norm_gate_kernel<<<ROWS * NVH, 128, 0, stream>>>(coref, zb, norm_w, yb);

  // 8) out = y @ w_out   (8192 x 2048 x 4096), fp32 output
  gemm_kernel<false><<<dim3(ROWS / 128, HID / 128), 256, 0, stream>>>(
      yb, woutb, (void*)out, ROWS, HID, VAL_DIM);
}
